// PreNormedHGTCavAttention_58239756534116
// MI455X (gfx1250) — matrix operand order repaired and checked by reading, not compiled, not run on hardware
//
#include <hip/hip_runtime.h>

typedef unsigned short u16;
typedef unsigned int   u32;

typedef __attribute__((ext_vector_type(16))) __bf16 v16bf;
typedef __attribute__((ext_vector_type(8)))  float  v8f;
typedef __attribute__((ext_vector_type(4)))  u32    u32x4;
typedef __attribute__((ext_vector_type(2)))  u32    u32x2;
typedef __attribute__((ext_vector_type(4)))  float  f32x4;
typedef __attribute__((ext_vector_type(8)))  int    i32x8;

union BFrag { u32x4 q[2]; v16bf v; };   // 32 bytes = one 16-lane bf16 WMMA fragment

#define B_  4
#define L_  5
#define HW_ 1024
#define C_  256
#define M_  8
#define D_  32
#define T_  2
#define R_  4
#define I_  256

__device__ __forceinline__ u16 f2bf(float f) {
  u32 u = __float_as_uint(f);
  u += 0x7FFFu + ((u >> 16) & 1u);          // round-to-nearest-even
  return (u16)(u >> 16);
}
__device__ __forceinline__ float bf2f(u16 h) { return __uint_as_float(((u32)h) << 16); }

__device__ __forceinline__ v8f wmma_bf16(BFrag a, BFrag b, v8f c) {
  return __builtin_amdgcn_wmma_f32_16x16x32_bf16(false, a.v, false, b.v, (short)0, c,
                                                 false, false);
}

// Issue one TDM DMA of `elems2B` bf16 elements from contiguous global memory
// `src` into LDS at `ldsAddr`; caller must s_wait_tensorcnt + barrier.
// Descriptor per CDNA5 ISA 8.3/8.4: 1-D tile, data_size = 2 bytes.
__device__ __forceinline__ void tdm_load_1d(u32 ldsAddr, const void* src, u32 elems2B) {
  unsigned long long ga = (unsigned long long)(size_t)src;
  u32x4 g0;
  g0.x = 1u;                                            // count=1, user descriptor
  g0.y = ldsAddr;                                       // lds_addr[31:0]
  g0.z = (u32)ga;                                       // global_addr[31:0]
  g0.w = ((u32)(ga >> 32) & 0x01FFFFFFu) | (2u << 30);  // addr[56:32] | type=2
  i32x8 g1;
  g1[0] = 0x10000;                       // workgroup_mask=0, data_size=1 (2 bytes)
  g1[1] = (int)(elems2B << 16);          // tensor_dim0[15:0] in bits[63:48]
  g1[2] = (int)((elems2B >> 16) | (1u << 16)); // tensor_dim0[31:16], tensor_dim1=1 (lo)
  g1[3] = (int)(elems2B << 16);          // tensor_dim1 hi=0, tile_dim0 = elems
  g1[4] = 1;                             // tile_dim1 = 1, tile_dim2 = 0
  g1[5] = (int)elems2B;                  // tensor_dim0_stride[31:0]
  g1[6] = 0;                             // stride0 hi, stride1 lo
  g1[7] = 0;
  asm volatile("tensor_load_to_lds %0, %1" :: "s"(g0), "s"(g1) : "memory");
}

// ---------------------------------------------------------------------------
// Kernel 0: bf16-convert + transpose weights; extract per-(b,l) types.
//  WqT/WkT/WvT : [T][I][C]  (row = output col, contiguous K=C)   from [T][C][I]
//  WaT         : [T][C][I]  (row = output col, contiguous K=I)   from [T][I][C]
//  rAT         : [R][M][q][p] (contiguous K=p)                   from [R][M][p][q]
//  rMT         : [R][M][c][p] (contiguous K=p)                   from [R][M][p][c]
// ---------------------------------------------------------------------------
__global__ void hgt_prep(const float* __restrict__ Wq, const float* __restrict__ Wk,
                         const float* __restrict__ Wv, const float* __restrict__ Wa,
                         const float* __restrict__ rA, const float* __restrict__ rMsg,
                         const float* __restrict__ prior,
                         u16* __restrict__ WqT, u16* __restrict__ WkT,
                         u16* __restrict__ WvT, u16* __restrict__ WaT,
                         u16* __restrict__ rAT, u16* __restrict__ rMT,
                         int* __restrict__ types)
{
  const int n1   = T_ * C_ * I_;          // 131072
  const int nRel = R_ * M_ * D_ * D_;     // 32768
  const int total = 4 * n1 + 2 * nRel + B_ * L_;
  for (int id = blockIdx.x * blockDim.x + threadIdx.x; id < total;
       id += gridDim.x * blockDim.x) {
    if (id < 3 * n1) {
      const float* W = (id < n1) ? Wq : (id < 2 * n1) ? Wk : Wv;
      u16* WT        = (id < n1) ? WqT : (id < 2 * n1) ? WkT : WvT;
      int r = id % n1;
      int t = r / (I_ * C_);
      int rem = r % (I_ * C_);
      int i = rem / C_, c = rem % C_;
      WT[r] = f2bf(W[(size_t)(t * C_ + c) * I_ + i]);
    } else if (id < 4 * n1) {
      int r = id - 3 * n1;
      int t = r / (C_ * I_);
      int rem = r % (C_ * I_);
      int c = rem / I_, i = rem % I_;
      WaT[r] = f2bf(Wa[(size_t)(t * I_ + i) * C_ + c]);
    } else if (id < 4 * n1 + nRel) {
      int r = id - 4 * n1;
      int rm = r / (D_ * D_);
      int rem = r % (D_ * D_);
      int qq = rem / D_, p = rem % D_;
      rAT[r] = f2bf(rA[(size_t)(rm * D_ + p) * D_ + qq]);
    } else if (id < 4 * n1 + 2 * nRel) {
      int r = id - 4 * n1 - nRel;
      int rm = r / (D_ * D_);
      int rem = r % (D_ * D_);
      int c = rem / D_, p = rem % D_;
      rMT[r] = f2bf(rMsg[(size_t)(rm * D_ + p) * D_ + c]);
    } else {
      int r = id - 4 * n1 - 2 * nRel;     // r = b*L + l
      types[r] = (int)prior[(size_t)r * (32 * 32 * 3) + 2];
    }
  }
}

// ---------------------------------------------------------------------------
// Kernel 1: LayerNorm over C=256, one wave per position; bf16 output
// xn layout [B][L][HW][C] == flat order of x [B,L,HS,WS,C].
// ---------------------------------------------------------------------------
__global__ void hgt_layernorm(const float* __restrict__ x, const float* __restrict__ lnw,
                              const float* __restrict__ lnb, u16* __restrict__ xn)
{
  int lane = threadIdx.x & 31;
  int wv   = threadIdx.x >> 5;
  int p    = blockIdx.x * 8 + wv;                 // 0 .. 20479 (exact)
  const float* row = x + (size_t)p * C_;

  f32x4 a0 = *(const f32x4*)(row + lane * 4);
  f32x4 a1 = *(const f32x4*)(row + 128 + lane * 4);
  float s = a0.x + a0.y + a0.z + a0.w + a1.x + a1.y + a1.z + a1.w;
#pragma unroll
  for (int off = 16; off >= 1; off >>= 1) s += __shfl_xor(s, off, 32);
  float mu = s * (1.0f / 256.0f);

  float d0 = a0.x - mu, d1 = a0.y - mu, d2 = a0.z - mu, d3 = a0.w - mu;
  float d4 = a1.x - mu, d5 = a1.y - mu, d6 = a1.z - mu, d7 = a1.w - mu;
  float vs = d0*d0 + d1*d1 + d2*d2 + d3*d3 + d4*d4 + d5*d5 + d6*d6 + d7*d7;
#pragma unroll
  for (int off = 16; off >= 1; off >>= 1) vs += __shfl_xor(vs, off, 32);
  float inv = rsqrtf(vs * (1.0f / 256.0f) + 1e-5f);

  f32x4 w0 = *(const f32x4*)(lnw + lane * 4);
  f32x4 w1 = *(const f32x4*)(lnw + 128 + lane * 4);
  f32x4 b0 = *(const f32x4*)(lnb + lane * 4);
  f32x4 b1 = *(const f32x4*)(lnb + 128 + lane * 4);

  u32x2 o0, o1;
  o0.x = (u32)f2bf(d0 * inv * w0.x + b0.x) | ((u32)f2bf(d1 * inv * w0.y + b0.y) << 16);
  o0.y = (u32)f2bf(d2 * inv * w0.z + b0.z) | ((u32)f2bf(d3 * inv * w0.w + b0.w) << 16);
  o1.x = (u32)f2bf(d4 * inv * w1.x + b1.x) | ((u32)f2bf(d5 * inv * w1.y + b1.y) << 16);
  o1.y = (u32)f2bf(d6 * inv * w1.z + b1.z) | ((u32)f2bf(d7 * inv * w1.w + b1.w) << 16);
  *(u32x2*)(xn + (size_t)p * C_ + lane * 4)       = o0;
  *(u32x2*)(xn + (size_t)p * C_ + 128 + lane * 4) = o1;
}

// ---------------------------------------------------------------------------
// Kernel 2: Q/K/V projections.  Block = 8 waves sharing a 64-col weight panel
// (32 KB LDS, loaded by one TDM tensor_load_to_lds).  Wave = 32x64 output tile:
// 2 A-frags x 4 B-frags -> 8 WMMAs per K-step (0.5 global + 1 LDS b128/WMMA).
// Output layout [B][M][L][HW][D] bf16 (head-major for the attention stage).
// ---------------------------------------------------------------------------
__global__ void hgt_qkv(const u16* __restrict__ xn,
                        const u16* __restrict__ WqT, const u16* __restrict__ WkT,
                        const u16* __restrict__ WvT,
                        const float* __restrict__ bq, const float* __restrict__ bk,
                        const float* __restrict__ bv,
                        const int* __restrict__ types,
                        u16* __restrict__ q, u16* __restrict__ k, u16* __restrict__ v)
{
  __shared__ __align__(16) u16 panel[64 * C_];    // 32 KB: 64 output cols x K=256
  int blk = blockIdx.x;                           // 20*3*4*4 = 960
  int rowGroup = blk & 3;  blk >>= 2;
  int colGroup = blk & 3;  blk >>= 2;
  int mat = blk % 3;
  int bl  = blk / 3;                              // b*L + l
  int type = types[bl];

  const u16* WT = (mat == 0 ? WqT : (mat == 1 ? WkT : WvT)) +
                  ((size_t)type * I_ + colGroup * 64) * C_;
  if (threadIdx.x < 32) {                         // wave 0 drives the TDM
    tdm_load_1d((u32)(size_t)(void*)panel, WT, 64u * C_);
    __builtin_amdgcn_s_wait_tensorcnt(0);
  }
  __syncthreads();

  int wv = threadIdx.x >> 5, lane = threadIdx.x & 31;
  int hh = lane >> 4, n = lane & 15;              // half-wave, lane-in-half
  int r0 = rowGroup * 256 + wv * 32;
  const u16* arow0 = xn + ((size_t)bl * HW_ + r0 + n) * C_;
  const u16* arow1 = arow0 + (size_t)16 * C_;

  v8f acc[8];
#pragma unroll
  for (int t = 0; t < 8; ++t) acc[t] = {};

#pragma unroll
  for (int k0 = 0; k0 < C_; k0 += 32) {
    BFrag a0, a1;
    a0.q[0] = *(const u32x4*)(arow0 + k0 + hh * 16);
    a0.q[1] = *(const u32x4*)(arow0 + k0 + 8 + hh * 16);
    a1.q[0] = *(const u32x4*)(arow1 + k0 + hh * 16);
    a1.q[1] = *(const u32x4*)(arow1 + k0 + 8 + hh * 16);
    if (k0 + 32 < C_) __builtin_prefetch(arow0 + k0 + 32, 0, 0);
#pragma unroll
    for (int ct = 0; ct < 4; ++ct) {
      BFrag bfr;
      const u16* bp = panel + (ct * 16 + n) * C_ + k0 + hh * 16;
      bfr.q[0] = *(const u32x4*)bp;
      bfr.q[1] = *(const u32x4*)(bp + 8);
      acc[ct]     = wmma_bf16(a0, bfr, acc[ct]);
      acc[4 + ct] = wmma_bf16(a1, bfr, acc[4 + ct]);
    }
  }

  const float* bias = (mat == 0 ? bq : (mat == 1 ? bk : bv)) + (size_t)type * I_;
  int b = bl / L_, l = bl % L_;
  u16* outp = (mat == 0 ? q : (mat == 1 ? k : v));
#pragma unroll
  for (int ct = 0; ct < 4; ++ct) {
    int col = colGroup * 64 + ct * 16 + n;
    float bia = bias[col];
    int m = col >> 5, d = col & 31;
    size_t obase = ((((size_t)b * M_ + m) * L_ + l) * HW_) * D_ + d;
#pragma unroll
    for (int rb = 0; rb < 2; ++rb) {
#pragma unroll
      for (int r = 0; r < 8; ++r) {
        int ro = r0 + rb * 16 + hh * 8 + r;        // C/D frag: M = h*8+r
        outp[obase + (size_t)ro * D_] = f2bf(acc[rb * 4 + ct][r] + bia);
      }
    }
  }
}

// ---------------------------------------------------------------------------
// Kernel 3: relation-indexed attention.  One wave per (b, m, 16-row tile, i):
// B*M*64*L = 10240 waves for latency hiding (the per-i work is independent and
// k/v/mask traffic is L2-resident either way).
// logits(i,j) = scale * rowdot(Q_i @ Watt[e], K_j);  softmax over j;
// out_i += att(i,j) * (V_j @ Wmsg[e]).
// ---------------------------------------------------------------------------
__global__ void hgt_attn(const u16* __restrict__ q, const u16* __restrict__ k,
                         const u16* __restrict__ v,
                         const u16* __restrict__ rAT, const u16* __restrict__ rMT,
                         const int* __restrict__ types, const int* __restrict__ mask,
                         u16* __restrict__ attout)
{
  int wv = threadIdx.x >> 5, lane = threadIdx.x & 31;
  int t  = blockIdx.x * 8 + wv;                   // 0..10239 (exact)
  int i  = t % L_;
  int u  = t / L_;                                // 0..2047
  int rt = u & 63;
  int bm = u >> 6;
  int m  = bm & (M_ - 1);
  int b  = bm >> 3;
  int hh = lane >> 4, n = lane & 15;
  int rowA = rt * 16 + n;
  const float scale = 0.17677669529663687f;       // 1/sqrt(32)

  int ti = types[b * L_ + i];
  const u16* qrow = q + ((((size_t)b * M_ + m) * L_ + i) * HW_ + rowA) * D_;
  BFrag qa;
  qa.q[0] = *(const u32x4*)(qrow + hh * 16);
  qa.q[1] = *(const u32x4*)(qrow + 8 + hh * 16);

  float lg[L_][8];
#pragma unroll
  for (int j = 0; j < L_; ++j) {
    int e = ti * T_ + types[b * L_ + j];
    const u16* wA = rAT + ((size_t)e * M_ + m) * (D_ * D_);
    BFrag b0, b1;
    const u16* bp0 = wA + n * D_ + hh * 16;           // N-tile 0
    b0.q[0] = *(const u32x4*)bp0;
    b0.q[1] = *(const u32x4*)(bp0 + 8);
    const u16* bp1 = wA + (16 + n) * D_ + hh * 16;    // N-tile 1
    b1.q[0] = *(const u32x4*)bp1;
    b1.q[1] = *(const u32x4*)(bp1 + 8);
    v8f z = {};
    v8f qw0 = wmma_bf16(qa, b0, z);
    v8f qw1 = wmma_bf16(qa, b1, z);

    const u16* kbase = k + ((((size_t)b * M_ + m) * L_ + j) * HW_) * D_;
    float pr[8];
#pragma unroll
    for (int r = 0; r < 8; ++r) {
      int row = rt * 16 + hh * 8 + r;
      float kv0 = bf2f(kbase[(size_t)row * D_ + n]);
      float kv1 = bf2f(kbase[(size_t)row * D_ + 16 + n]);
      pr[r] = qw0[r] * kv0 + qw1[r] * kv1;
    }
#pragma unroll
    for (int off = 8; off >= 1; off >>= 1) {          // reduce over 16 lanes/half
#pragma unroll
      for (int r = 0; r < 8; ++r) pr[r] += __shfl_xor(pr[r], off, 32);
    }
#pragma unroll
    for (int r = 0; r < 8; ++r) {
      int row = rt * 16 + hh * 8 + r;
      int mv = mask[(((size_t)b * HW_ + row) * L_ + i) * L_ + j];
      lg[j][r] = (mv == 0) ? -1e9f : pr[r] * scale;
    }
  }

  // softmax over j (register-resident)
#pragma unroll
  for (int r = 0; r < 8; ++r) {
    float mx = lg[0][r];
#pragma unroll
    for (int j = 1; j < L_; ++j) mx = fmaxf(mx, lg[j][r]);
    float s = 0.f;
#pragma unroll
    for (int j = 0; j < L_; ++j) { float e = __expf(lg[j][r] - mx); lg[j][r] = e; s += e; }
    float invs = 1.0f / s;
#pragma unroll
    for (int j = 0; j < L_; ++j) lg[j][r] *= invs;
  }

  // messages
  float acc0[8] = {0,0,0,0,0,0,0,0}, acc1[8] = {0,0,0,0,0,0,0,0};
#pragma unroll
  for (int j = 0; j < L_; ++j) {
    int e = ti * T_ + types[b * L_ + j];
    const u16* wM = rMT + ((size_t)e * M_ + m) * (D_ * D_);
    const u16* vrow = v + ((((size_t)b * M_ + m) * L_ + j) * HW_ + rowA) * D_;
    BFrag va, b0, b1;
    va.q[0] = *(const u32x4*)(vrow + hh * 16);
    va.q[1] = *(const u32x4*)(vrow + 8 + hh * 16);
    const u16* bp0 = wM + n * D_ + hh * 16;
    b0.q[0] = *(const u32x4*)bp0;
    b0.q[1] = *(const u32x4*)(bp0 + 8);
    const u16* bp1 = wM + (16 + n) * D_ + hh * 16;
    b1.q[0] = *(const u32x4*)bp1;
    b1.q[1] = *(const u32x4*)(bp1 + 8);
    v8f z = {};
    v8f d0 = wmma_bf16(va, b0, z);
    v8f d1 = wmma_bf16(va, b1, z);
#pragma unroll
    for (int r = 0; r < 8; ++r) {
      acc0[r] += lg[j][r] * d0[r];
      acc1[r] += lg[j][r] * d1[r];
    }
  }

  // store [B][L][HW][I] bf16, channel = m*32 + d
  size_t obase = ((size_t)b * L_ + i) * HW_ * C_ + m * D_;
#pragma unroll
  for (int r = 0; r < 8; ++r) {
    int row = rt * 16 + hh * 8 + r;
    attout[obase + (size_t)row * C_ + n]      = f2bf(acc0[r]);
    attout[obase + (size_t)row * C_ + 16 + n] = f2bf(acc1[r]);
  }
}

// ---------------------------------------------------------------------------
// Kernel 4: output projection -> f32 d_out in [B,L,HS,WS,C] order.
// Same 32x64 per-wave blocking; panel via cooperative ds-path copy.
// ---------------------------------------------------------------------------
__global__ void hgt_outproj(const u16* __restrict__ attout, const u16* __restrict__ WaT,
                            const float* __restrict__ ba, const int* __restrict__ types,
                            float* __restrict__ out)
{
  __shared__ __align__(16) u16 panel[64 * I_];    // 32 KB
  int blk = blockIdx.x;                           // 20*4*4 = 320
  int rowGroup = blk & 3;  blk >>= 2;
  int colGroup = blk & 3;  blk >>= 2;
  int bl = blk;
  int type = types[bl];
  const u16* WT = WaT + ((size_t)type * C_ + colGroup * 64) * I_;
  {
    const u32x4* src = (const u32x4*)WT;
    u32x4* dst = (u32x4*)panel;
#pragma unroll
    for (int t = 0; t < 8; ++t) dst[threadIdx.x + t * 256] = src[threadIdx.x + t * 256];
  }
  __syncthreads();

  int wv = threadIdx.x >> 5, lane = threadIdx.x & 31;
  int hh = lane >> 4, n = lane & 15;
  int r0 = rowGroup * 256 + wv * 32;
  const u16* arow0 = attout + ((size_t)bl * HW_ + r0 + n) * C_;
  const u16* arow1 = arow0 + (size_t)16 * C_;

  v8f acc[8];
#pragma unroll
  for (int t = 0; t < 8; ++t) acc[t] = {};

#pragma unroll
  for (int k0 = 0; k0 < I_; k0 += 32) {
    BFrag a0, a1;
    a0.q[0] = *(const u32x4*)(arow0 + k0 + hh * 16);
    a0.q[1] = *(const u32x4*)(arow0 + k0 + 8 + hh * 16);
    a1.q[0] = *(const u32x4*)(arow1 + k0 + hh * 16);
    a1.q[1] = *(const u32x4*)(arow1 + k0 + 8 + hh * 16);
    if (k0 + 32 < I_) __builtin_prefetch(arow0 + k0 + 32, 0, 0);
#pragma unroll
    for (int ct = 0; ct < 4; ++ct) {
      BFrag bfr;
      const u16* bp = panel + (ct * 16 + n) * I_ + k0 + hh * 16;
      bfr.q[0] = *(const u32x4*)bp;
      bfr.q[1] = *(const u32x4*)(bp + 8);
      acc[ct]     = wmma_bf16(a0, bfr, acc[ct]);
      acc[4 + ct] = wmma_bf16(a1, bfr, acc[4 + ct]);
    }
  }

#pragma unroll
  for (int ct = 0; ct < 4; ++ct) {
    int col = colGroup * 64 + ct * 16 + n;
    float bia = ba[(size_t)type * C_ + col];
    size_t obase = (size_t)bl * HW_ * C_ + col;
#pragma unroll
    for (int rb = 0; rb < 2; ++rb) {
#pragma unroll
      for (int r = 0; r < 8; ++r) {
        int ro = r0 + rb * 16 + hh * 8 + r;
        out[obase + (size_t)ro * C_] = acc[rb * 4 + ct][r] + bia;
      }
    }
  }
}

// ---------------------------------------------------------------------------
extern "C" void kernel_launch(void* const* d_in, const int* in_sizes, int n_in,
                              void* d_out, int out_size, void* d_ws, size_t ws_size,
                              hipStream_t stream)
{
  const float* x     = (const float*)d_in[0];
  const int*   maskp = (const int*)d_in[1];
  const float* prior = (const float*)d_in[2];
  const float* lnw   = (const float*)d_in[3];
  const float* lnb   = (const float*)d_in[4];
  const float* Wq    = (const float*)d_in[5];
  const float* bq    = (const float*)d_in[6];
  const float* Wk    = (const float*)d_in[7];
  const float* bk    = (const float*)d_in[8];
  const float* Wv    = (const float*)d_in[9];
  const float* bv    = (const float*)d_in[10];
  const float* Wa    = (const float*)d_in[11];
  const float* ba    = (const float*)d_in[12];
  const float* rA    = (const float*)d_in[13];
  const float* rM    = (const float*)d_in[14];
  (void)in_sizes; (void)n_in; (void)out_size; (void)ws_size;

  char* ws = (char*)d_ws;
  size_t off = 0;
  auto carve = [&](size_t bytes) -> void* {
    void* p = ws + off;
    off += (bytes + 255) & ~(size_t)255;
    return p;
  };
  u16* WqT    = (u16*)carve((size_t)T_ * I_ * C_ * 2);           // 256 KB
  u16* WkT    = (u16*)carve((size_t)T_ * I_ * C_ * 2);
  u16* WvT    = (u16*)carve((size_t)T_ * I_ * C_ * 2);
  u16* WaT    = (u16*)carve((size_t)T_ * C_ * I_ * 2);
  u16* rAT    = (u16*)carve((size_t)R_ * M_ * D_ * D_ * 2);      // 64 KB
  u16* rMT    = (u16*)carve((size_t)R_ * M_ * D_ * D_ * 2);
  int* typesW = (int*)carve((size_t)B_ * L_ * sizeof(int));
  u16* xn     = (u16*)carve((size_t)B_ * L_ * HW_ * C_ * 2);     // 10.5 MB
  u16* qb     = (u16*)carve((size_t)B_ * M_ * L_ * HW_ * D_ * 2);
  u16* kb     = (u16*)carve((size_t)B_ * M_ * L_ * HW_ * D_ * 2);
  u16* vb     = (u16*)carve((size_t)B_ * M_ * L_ * HW_ * D_ * 2);
  u16* ao     = (u16*)carve((size_t)B_ * L_ * HW_ * C_ * 2);     // total ~54 MB

  hipLaunchKernelGGL(hgt_prep, dim3(1024), dim3(256), 0, stream,
                     Wq, Wk, Wv, Wa, rA, rM, prior,
                     WqT, WkT, WvT, WaT, rAT, rMT, typesW);
  hipLaunchKernelGGL(hgt_layernorm, dim3(2560), dim3(256), 0, stream, x, lnw, lnb, xn);
  hipLaunchKernelGGL(hgt_qkv, dim3(960), dim3(256), 0, stream,
                     xn, WqT, WkT, WvT, bq, bk, bv, typesW, qb, kb, vb);
  hipLaunchKernelGGL(hgt_attn, dim3(1280), dim3(256), 0, stream,
                     qb, kb, vb, rAT, rMT, typesW, maskp, ao);
  hipLaunchKernelGGL(hgt_outproj, dim3(320), dim3(256), 0, stream,
                     ao, WaT, ba, typesW, (float*)d_out);
}
